// GRU_55233279426852
// MI455X (gfx1250) — matrix operand order repaired and checked by reading, not compiled, not run on hardware
//
#include <hip/hip_runtime.h>
#include <hip/hip_bf16.h>

// GRU (B=64, L=1024, I=256, H=512) as a single persistent CDNA5 kernel.
//
// 4 batch-groups x 16 rows; each group = 8 WGs splitting H into 64-col slices;
// each WG = 4 waves; a wave owns a 16x16 tile and keeps its h-state in the
// WMMA f32 C/D register layout across all 1024 steps. Weights live in LDS as
// bf16 (~318 KB of the 320 KB/WGP). Per step: stage x_t / h_{t-1} into LDS,
// 72x v_wmma_f32_16x16x32_bf16 per wave, fused GRU elementwise update,
// publish h_t (f32 to the output tensor + bf16 to a d_ws exchange ring),
// then an atomic-counter barrier among the 8 WGs (+ cluster barrier).
// h_{t-1} staging uses global_load_async_to_lds_b128 (ASYNCcnt path) when
// d_ws is large enough; otherwise falls back to f32 loads via the output.

typedef __bf16 bf16;
typedef __attribute__((ext_vector_type(16))) bf16  v16bf;
typedef __attribute__((ext_vector_type(8)))  bf16  v8bf;
typedef __attribute__((ext_vector_type(4)))  bf16  v4bf;
typedef __attribute__((ext_vector_type(8)))  float v8f;

#define B_   64
#define L_   1024
#define I_   256
#define H_   512
#define BB   16          // batch rows per group
#define GWG  8           // workgroups per group
#define NGRP (B_ / BB)   // 4 groups
#define CPW  64          // output columns per workgroup
#define HP   520         // padded H-K stride (bf16 elems): bank-conflict-free b128
#define IP   264         // padded I-K stride

// LDS layout (bf16 element offsets); dynamic LDS starts at offset 0
#define OFF_WHH 0
#define SZ_WHH  (3 * CPW * HP)          // 99840
#define OFF_WIH (OFF_WHH + SZ_WHH)
#define SZ_WIH  (3 * CPW * IP)          // 50688
#define OFF_HST (OFF_WIH + SZ_WIH)
#define SZ_HST  (BB * HP)               // 8320
#define OFF_XST (OFF_HST + SZ_HST)
#define SZ_XST  (BB * IP)               // 4224
#define LDS_ELEMS (OFF_XST + SZ_XST)    // 163072 elems
#define LDS_BYTES (LDS_ELEMS * 2)       // 326144 B  (<= 320 KB WGP LDS)

// d_ws layout: [0,16) barrier counters; exchange ring at +256
#define EXCH_OFF   256
#define EXCH_SLOT  (NGRP * BB * H_)                   // bf16 elems per ring slot
#define WS_NEEDED  (EXCH_OFF + 2 * EXCH_SLOT * 2)     // bytes (double buffer)

__device__ __forceinline__ float sigf(float x) {
    return __builtin_amdgcn_rcpf(1.0f + __expf(-x));
}
__device__ __forceinline__ float tanhfast(float x) {
    return 2.0f * __builtin_amdgcn_rcpf(1.0f + __expf(-2.0f * x)) - 1.0f;
}

// A fragment (16x32 bf16): lane l holds row l&15; K = {hi..hi+7, 16+hi..16+hi+7}
__device__ __forceinline__ v16bf load_a(const bf16* row, int hi, int k0) {
    v8bf lo = *(const v8bf*)(row + k0 + hi);
    v8bf hh = *(const v8bf*)(row + k0 + 16 + hi);
    v16bf a;
#pragma unroll
    for (int i = 0; i < 8; ++i) { a[i] = lo[i]; a[8 + i] = hh[i]; }
    return a;
}
// B fragment (32x16 bf16): lane l holds col l&15; K = {kh..kh+7, 16+kh..16+kh+7}
__device__ __forceinline__ v16bf load_b(const bf16* col, int kh, int k0) {
    v8bf lo = *(const v8bf*)(col + k0 + kh);
    v8bf hh = *(const v8bf*)(col + k0 + 16 + kh);
    v16bf b;
#pragma unroll
    for (int i = 0; i < 8; ++i) { b[i] = lo[i]; b[8 + i] = hh[i]; }
    return b;
}
__device__ __forceinline__ v8f wmma_bf16(v16bf a, v16bf b, v8f c) {
    return __builtin_amdgcn_wmma_f32_16x16x32_bf16(
        false, a, false, b, (short)0, c, false, false);
}

__global__ void init_counters(unsigned* cnt) {
    if (threadIdx.x < NGRP) cnt[threadIdx.x] = 0u;
}

template <bool USE_EXCH>
__global__ void __launch_bounds__(128, 1)
gru_persistent(const float* __restrict__ inp, const float* __restrict__ h0,
               const float* __restrict__ wir, const float* __restrict__ wiz,
               const float* __restrict__ win, const float* __restrict__ bir,
               const float* __restrict__ biz, const float* __restrict__ binn,
               const float* __restrict__ whr, const float* __restrict__ whz,
               const float* __restrict__ whn, const float* __restrict__ bhr,
               const float* __restrict__ bhz, const float* __restrict__ bhn,
               float* __restrict__ out, unsigned* __restrict__ cnt,
               bf16* __restrict__ exch)
{
    extern __shared__ bf16 sh[];
    bf16* s_whh = sh + OFF_WHH;   // [3][CPW][HP]
    bf16* s_wih = sh + OFF_WIH;   // [3][CPW][IP]
    bf16* s_h   = sh + OFF_HST;   // [BB][HP]
    bf16* s_x   = sh + OFF_XST;   // [BB][IP]

    const int tid  = threadIdx.x;
    const int wave = tid >> 5;
    const int lane = tid & 31;
    const int grp  = blockIdx.x >> 3;       // batch group 0..3
    const int b0   = grp * BB;
    const int col0 = (blockIdx.x & 7) * CPW;

    // ---- preload weight slices (f32 -> bf16) into LDS ----
    const float* wh[3] = { whr, whz, whn };
    const float* wi[3] = { wir, wiz, win };
#pragma unroll
    for (int g = 0; g < 3; ++g) {
        const float* src = wh[g];
        bf16* dst = s_whh + g * CPW * HP;
        for (int idx = tid * 4; idx < CPW * H_; idx += 128 * 4) {
            int cc = idx >> 9, k = idx & (H_ - 1);
            float4 f = *(const float4*)(src + (size_t)(col0 + cc) * H_ + k);
            v4bf p; p[0] = (bf16)f.x; p[1] = (bf16)f.y; p[2] = (bf16)f.z; p[3] = (bf16)f.w;
            *(v4bf*)(dst + cc * HP + k) = p;
        }
        const float* si = wi[g];
        bf16* di = s_wih + g * CPW * IP;
        for (int idx = tid * 4; idx < CPW * I_; idx += 128 * 4) {
            int cc = idx >> 8, k = idx & (I_ - 1);
            float4 f = *(const float4*)(si + (size_t)(col0 + cc) * I_ + k);
            v4bf p; p[0] = (bf16)f.x; p[1] = (bf16)f.y; p[2] = (bf16)f.z; p[3] = (bf16)f.w;
            *(v4bf*)(di + cc * IP + k) = p;
        }
    }

    // ---- per-lane constants ----
    const int nn_  = lane & 15;
    const int c    = col0 + wave * 16 + nn_;    // global output column (C/D lane col)
    const int hiA  = (lane >> 4) << 3;          // A-frag K sub-offset: 0 or 8
    const int khB  = (lane >> 4) << 3;          // B-frag K sub-offset: 0 or 8
    const int radd = (lane >> 4) << 3;          // C/D row add: 0 or 8
    const float bR = bir[c] + bhr[c];
    const float bZ = biz[c] + bhz[c];
    const float bN = binn[c];
    const float bH = bhn[c];

    const int jl = wave * 16 + nn_;             // local column in WG slice
    const bf16* aH  = s_h + (lane & 15) * HP;
    const bf16* aX  = s_x + (lane & 15) * IP;
    const bf16* bHr = s_whh + 0 * CPW * HP + jl * HP;
    const bf16* bHz = s_whh + 1 * CPW * HP + jl * HP;
    const bf16* bHn = s_whh + 2 * CPW * HP + jl * HP;
    const bf16* bXr = s_wih + 0 * CPW * IP + jl * IP;
    const bf16* bXz = s_wih + 1 * CPW * IP + jl * IP;
    const bf16* bXn = s_wih + 2 * CPW * IP + jl * IP;

    // h state in WMMA C/D register layout
    v8f hreg;
#pragma unroll
    for (int v = 0; v < 8; ++v)
        hreg[v] = h0[(size_t)(b0 + radd + v) * H_ + c];

    __syncthreads();   // weights visible to whole WG

    for (int t = 0; t < L_; ++t) {
        // ---- stage x_t (f32 -> bf16) into LDS ----
        for (int idx = tid * 4; idx < BB * I_; idx += 128 * 4) {
            int r = idx >> 8, k = idx & (I_ - 1);
            float4 f = *(const float4*)(inp + ((size_t)(b0 + r) * L_ + t) * I_ + k);
            v4bf p; p[0] = (bf16)f.x; p[1] = (bf16)f.y; p[2] = (bf16)f.z; p[3] = (bf16)f.w;
            *(v4bf*)(s_x + r * IP + k) = p;
        }
        // ---- stage h_{t-1} into LDS ----
        if (USE_EXCH && t > 0) {
            // bf16 exchange ring -> LDS via the async data mover (ASYNCcnt).
            // 16 rows x 512 bf16 = 1024 x 16B chunks; 8 per thread.
            const bf16* ebase = exch + (size_t)((t - 1) & 1) * EXCH_SLOT
                                     + (size_t)grp * (BB * H_);
            for (int idx = tid; idx < (BB * H_) / 8; idx += 128) {
                unsigned go = (unsigned)idx * 16u;                         // global byte off
                unsigned lo = (unsigned)(OFF_HST * 2) +
                              (unsigned)(idx >> 6) * (unsigned)(HP * 2) +
                              (unsigned)(idx & 63) * 16u;                  // LDS byte off
                asm volatile("global_load_async_to_lds_b128 %0, %1, %2"
                             :: "v"(lo), "v"(go), "s"(ebase) : "memory");
            }
            asm volatile("s_wait_asynccnt 0" ::: "memory");
        } else {
            const float* hsrc = (t == 0) ? (h0 + (size_t)b0 * H_)
                                         : (out + ((size_t)b0 * L_ + (t - 1)) * H_);
            const size_t hstr = (t == 0) ? (size_t)H_ : (size_t)L_ * H_;
            for (int idx = tid * 4; idx < BB * H_; idx += 128 * 4) {
                int r = idx >> 9, k = idx & (H_ - 1);
                float4 f = *(const float4*)(hsrc + (size_t)r * hstr + k);
                v4bf p; p[0] = (bf16)f.x; p[1] = (bf16)f.y; p[2] = (bf16)f.z; p[3] = (bf16)f.w;
                *(v4bf*)(s_h + r * HP + k) = p;
            }
        }
        __syncthreads();

        if (wave == 0 && lane < BB && t + 1 < L_)
            __builtin_prefetch(inp + ((size_t)(b0 + lane) * L_ + (t + 1)) * I_, 0, 3);

        // ---- 3-gate GEMMs: x-part (K=256) then h-part (K=512) ----
        v8f accR = {}, accZ = {}, accNX = {}, accNH = {};
#pragma unroll 4
        for (int k0 = 0; k0 < I_; k0 += 32) {
            v16bf a = load_a(aX, hiA, k0);
            accR  = wmma_bf16(a, load_b(bXr, khB, k0), accR);
            accZ  = wmma_bf16(a, load_b(bXz, khB, k0), accZ);
            accNX = wmma_bf16(a, load_b(bXn, khB, k0), accNX);
        }
#pragma unroll 4
        for (int k0 = 0; k0 < H_; k0 += 32) {
            v16bf a = load_a(aH, hiA, k0);
            accR  = wmma_bf16(a, load_b(bHr, khB, k0), accR);
            accZ  = wmma_bf16(a, load_b(bHz, khB, k0), accZ);
            accNH = wmma_bf16(a, load_b(bHn, khB, k0), accNH);
        }

        // ---- GRU elementwise update; publish h_t ----
        bf16* edst = USE_EXCH
                   ? (exch + (size_t)(t & 1) * EXCH_SLOT + (size_t)grp * (BB * H_))
                   : (bf16*)nullptr;
#pragma unroll
        for (int v = 0; v < 8; ++v) {
            float rr = sigf(accR[v] + bR);
            float zz = sigf(accZ[v] + bZ);
            float nv = tanhfast(accNX[v] + bN + rr * (accNH[v] + bH));
            float hn = (1.0f - zz) * nv + zz * hreg[v];
            hreg[v] = hn;
            out[((size_t)(b0 + radd + v) * L_ + t) * H_ + c] = hn;
            if (USE_EXCH)
                edst[(size_t)(radd + v) * H_ + c] = (bf16)hn;
        }

        // ---- barrier among the 8 WGs of this batch group ----
        __threadfence();
        __syncthreads();
        if (tid == 0) {
            __hip_atomic_fetch_add(&cnt[grp], 1u, __ATOMIC_ACQ_REL,
                                   __HIP_MEMORY_SCOPE_AGENT);
            const unsigned tgt = (unsigned)GWG * (unsigned)(t + 1);
            while (__hip_atomic_load(&cnt[grp], __ATOMIC_ACQUIRE,
                                     __HIP_MEMORY_SCOPE_AGENT) < tgt)
                __builtin_amdgcn_s_sleep(2);
        }
        __builtin_amdgcn_s_cluster_barrier();  // NOP unclustered; HW path when clustered
        __syncthreads();
        __threadfence();
    }

    // ---- final hidden state (second output, concatenated after hs) ----
    float* hlast = out + (size_t)B_ * L_ * H_;
#pragma unroll
    for (int v = 0; v < 8; ++v)
        hlast[(size_t)(b0 + radd + v) * H_ + c] = hreg[v];
}

extern "C" void kernel_launch(void* const* d_in, const int* in_sizes, int n_in,
                              void* d_out, int out_size, void* d_ws, size_t ws_size,
                              hipStream_t stream) {
    (void)in_sizes; (void)n_in; (void)out_size;
    const float* inp = (const float*)d_in[0];
    const float* h0  = (const float*)d_in[1];
    const float* wir = (const float*)d_in[2];
    const float* wiz = (const float*)d_in[3];
    const float* win = (const float*)d_in[4];
    const float* bir = (const float*)d_in[5];
    const float* biz = (const float*)d_in[6];
    const float* bin = (const float*)d_in[7];
    const float* whr = (const float*)d_in[8];
    const float* whz = (const float*)d_in[9];
    const float* whn = (const float*)d_in[10];
    const float* bhr = (const float*)d_in[11];
    const float* bhz = (const float*)d_in[12];
    const float* bhn = (const float*)d_in[13];
    float* out    = (float*)d_out;
    unsigned* cnt = (unsigned*)d_ws;
    bf16* exch    = (bf16*)((char*)d_ws + EXCH_OFF);

    init_counters<<<1, 64, 0, stream>>>(cnt);
    if (ws_size >= (size_t)WS_NEEDED) {
        gru_persistent<true><<<dim3(GWG * NGRP), dim3(128), LDS_BYTES, stream>>>(
            inp, h0, wir, wiz, win, bir, biz, bin,
            whr, whz, whn, bhr, bhz, bhn, out, cnt, exch);
    } else {
        gru_persistent<false><<<dim3(GWG * NGRP), dim3(128), LDS_BYTES, stream>>>(
            inp, h0, wir, wiz, win, bir, biz, bin,
            whr, whz, whn, bhr, bhz, bhn, out, cnt, exch);
    }
}
